// TSDFVol_79706003079354
// MI455X (gfx1250) — matrix operand order repaired and checked by reading, not compiled, not run on hardware
//
#include <hip/hip_runtime.h>
#include <cstdint>
#include <cstddef>

typedef _Float16 h16;
typedef __attribute__((ext_vector_type(16))) _Float16 v16h;
typedef __attribute__((ext_vector_type(8)))  float    v8f;
typedef __attribute__((ext_vector_type(4)))  unsigned v4u;

#define SA 288
#define SB 288
#define SC 320

// packed f16 weight offsets (in halfs), each matrix padded to [Opad][Kpad]
#define OFF_DW1 0        // 256 x 64
#define OFF_DW2 16384    // 256 x 256
#define OFF_DW3 81920    // 272 x 256 (257 valid rows)
#define OFF_CW1 151552   // 256 x 320 (318 valid cols)
#define OFF_CW2 233472   // 256 x 256
#define OFF_CW3 299008   // 16  x 256 (3 valid rows)
#define OFF_TW1 303104   // 64  x 32  (2 valid cols)
#define OFF_TW2 305152   // 128 x 64
#define OFF_TW3 313344   // 128 x 128
#define OFF_TW4 329728   // 64  x 128
#define OFF_TW5 337920   // 16  x 64  (2 valid rows)
#define WPACK_TOTAL 338944

__device__ __forceinline__ int imin(int a, int b) { return a < b ? a : b; }
__device__ __forceinline__ int imax(int a, int b) { return a > b ? a : b; }

// ---------------- weight packing: f32 [O][I] -> f16 [Opad][Ipad], zero-filled ----
struct PackDesc { const float* src; unsigned dst_off; int O, I, Opad, Ipad; };
struct PackArgs { PackDesc d[11]; };

__global__ void pack_weights_kernel(PackArgs pa, h16* __restrict__ dst) {
  unsigned tid = blockIdx.x * blockDim.x + threadIdx.x;
  unsigned off = 0;
  for (int m = 0; m < 11; ++m) {
    unsigned sz = (unsigned)pa.d[m].Opad * (unsigned)pa.d[m].Ipad;
    if (tid < off + sz) {
      unsigned li = tid - off;
      int i = (int)(li % (unsigned)pa.d[m].Ipad);
      int o = (int)(li / (unsigned)pa.d[m].Ipad);
      float v = 0.0f;
      if (o < pa.d[m].O && i < pa.d[m].I)
        v = pa.d[m].src[(size_t)o * pa.d[m].I + i];
      dst[pa.d[m].dst_off + li] = (h16)v;
      return;
    }
    off += sz;
  }
}

// ---------------- trilinear helpers ------------------------------------------
__device__ __forceinline__ void triSetup(float px, float py, float pz, int N,
                                         int& x0, int& y0, int& z0,
                                         float& fx, float& fy, float& fz) {
  float t, fl; int i;
  t = fminf(fmaxf((px + 1.0f) * 0.5f, 0.0f), 1.0f) * (float)(N - 1);
  fl = floorf(t); i = imin(imax((int)fl, 0), N - 2); fx = t - (float)i; x0 = i;
  t = fminf(fmaxf((py + 1.0f) * 0.5f, 0.0f), 1.0f) * (float)(N - 1);
  fl = floorf(t); i = imin(imax((int)fl, 0), N - 2); fy = t - (float)i; y0 = i;
  t = fminf(fmaxf((pz + 1.0f) * 0.5f, 0.0f), 1.0f) * (float)(N - 1);
  fl = floorf(t); i = imin(imax((int)fl, 0), N - 2); fz = t - (float)i; z0 = i;
}

__device__ __forceinline__ float triSample(const float* __restrict__ g, int N, int c,
                                           int x0, int y0, int z0,
                                           float fx, float fy, float fz) {
  size_t N2 = (size_t)N * N;
  const float* G = g + (size_t)c * N * N2;
  size_t i000 = (size_t)x0 * N2 + (size_t)y0 * N + (size_t)z0;
  float g000 = G[i000],        g001 = G[i000 + 1];
  float g010 = G[i000 + N],    g011 = G[i000 + N + 1];
  float g100 = G[i000 + N2],   g101 = G[i000 + N2 + 1];
  float g110 = G[i000 + N2 + N], g111 = G[i000 + N2 + N + 1];
  float c00 = g000 * (1.0f - fz) + g001 * fz;
  float c01 = g010 * (1.0f - fz) + g011 * fz;
  float c10 = g100 * (1.0f - fz) + g101 * fz;
  float c11 = g110 * (1.0f - fz) + g111 * fz;
  float c0 = c00 * (1.0f - fy) + c01 * fy;
  float c1 = c10 * (1.0f - fy) + c11 * fy;
  return c0 * (1.0f - fx) + c1 * fx;
}

// ---------------- per-wave 16-point GEMM layer via v_wmma_f32_16x16x32_f16 ----
// actIn  (LDS): [32 points][inS halfs], row p, features 0..Kpad-1 (zero padded)
// actOut (LDS): [32 points][outS halfs], features 0..Opad-1
// Wh (global packed f16): [Opad][Kpad] row-major
__device__ __forceinline__ void wave_gemm(
    const h16* __restrict__ actIn, int inS,
    h16* __restrict__ actOut, int outS,
    const h16* __restrict__ Wh, const float* __restrict__ bias,
    int Kpad, int Opad, int Ovalid, bool relu,
    int tp, int lane) {
  const int l16 = lane & 15;
  const int g   = lane >> 4;
  const int KT  = Kpad >> 5;
  const int NT  = Opad >> 4;
  const h16* arow = actIn + (size_t)(tp + l16) * inS + g * 8;
  for (int nt = 0; nt < NT; ++nt) {
    const int ofeat = nt * 16 + l16;
    float bv = (ofeat < Ovalid) ? bias[ofeat] : 0.0f;
    v8f c = { bv, bv, bv, bv, bv, bv, bv, bv };
    const h16* wrow = Wh + (size_t)ofeat * Kpad + g * 8;
    for (int kt = 0; kt < KT; ++kt) {
      const int kb = kt << 5;
      union { v4u u[2]; v16h v; } A, B;
      // A layout: K(e) = kb + e + 8*g (e<8) ; kb + 16 + (e-8) + 8*g (e>=8)
      A.u[0] = *(const v4u*)(arow + kb);
      A.u[1] = *(const v4u*)(arow + kb + 16);
      // B layout: K(e) = kb + e + 8*g (e<8) ; kb + 16 + (e-8) + 8*g (e>=8) ; N = l16
      B.u[0] = *(const v4u*)(wrow + kb);
      B.u[1] = *(const v4u*)(wrow + kb + 16);
      c = __builtin_amdgcn_wmma_f32_16x16x32_f16(false, A.v, false, B.v,
                                                 (short)0, c, false, false);
    }
#pragma unroll
    for (int v = 0; v < 8; ++v) {
      float x = c[v];
      if (relu) x = x > 0.0f ? x : 0.0f;
      // C layout: point = v + 8*g, feature = nt*16 + l16
      actOut[(size_t)(tp + v + 8 * g) * outS + nt * 16 + l16] = (h16)x;
    }
  }
}

// ---------------- fused per-point kernel: 32 points / block (2 waves) --------
__global__ void __launch_bounds__(64) tsdfvol_fused_kernel(
    const float* __restrict__ xyz, const float* __restrict__ viewdirs,
    const float* __restrict__ tsdf_grid, const float* __restrict__ feat_grid,
    const float* __restrict__ color_grid,
    const h16* __restrict__ wp,
    const float* __restrict__ db1, const float* __restrict__ db2,
    const float* __restrict__ db3,
    const float* __restrict__ cb1, const float* __restrict__ cb2,
    const float* __restrict__ cb3,
    const float* __restrict__ tb1, const float* __restrict__ tb2,
    const float* __restrict__ tb3, const float* __restrict__ tb4,
    const float* __restrict__ tb5,
    float* __restrict__ sigOut, float* __restrict__ rgbOut) {
  __shared__ h16 bufA[32 * SA];
  __shared__ h16 bufB[32 * SB];
  __shared__ h16 bufC[32 * SC];
  __shared__ float tsdfL[32];
  __shared__ float sigmaL[32];

  const int t    = threadIdx.x;
  const int lane = t & 31;
  const int wave = t >> 5;
  const int tp   = wave * 16;
  const int l16  = lane & 15;

  // ---- Phase 1: trilinear gathers + input assembly (2 threads / point) ----
  {
    const int p  = t >> 1;
    const int hf = t & 1;
    const int gp = blockIdx.x * 32 + p;
    const float px = xyz[gp * 3 + 0], py = xyz[gp * 3 + 1], pz = xyz[gp * 3 + 2];
    const int ray = gp >> 7;  // S = 128
    float vx = viewdirs[ray * 3 + 0], vy = viewdirs[ray * 3 + 1], vz = viewdirs[ray * 3 + 2];
    const float inv = rsqrtf(vx * vx + vy * vy + vz * vz);
    vx *= inv; vy *= inv; vz *= inv;

    int x0, y0, z0; float fx, fy, fz;
    triSetup(px, py, pz, 128, x0, y0, z0, fx, fy, fz);
    for (int c = hf * 16; c < hf * 16 + 16; ++c) {
      bufA[p * SA + 3 + c]   = (h16)triSample(feat_grid, 128, c, x0, y0, z0, fx, fy, fz);
      bufC[p * SC + 259 + c] = (h16)triSample(color_grid, 128, c, x0, y0, z0, fx, fy, fz);
    }
    if (hf == 0) {
      int a0, b0, c0; float gx, gy, gz;
      triSetup(px, py, pz, 256, a0, b0, c0, gx, gy, gz);
      tsdfL[p] = triSample(tsdf_grid, 256, 0, a0, b0, c0, gx, gy, gz);
      bufA[p * SA + 0] = (h16)px; bufA[p * SA + 1] = (h16)py; bufA[p * SA + 2] = (h16)pz;
      for (int j = 35; j < 64; ++j) bufA[p * SA + j] = (h16)0.0f;
      bufC[p * SC + 0] = (h16)px; bufC[p * SC + 1] = (h16)py; bufC[p * SC + 2] = (h16)pz;
      bufC[p * SC + 291] = (h16)vx; bufC[p * SC + 292] = (h16)vy; bufC[p * SC + 293] = (h16)vz;
      const float pc[3] = { px, py, pz }, vc[3] = { vx, vy, vz };
      for (int k = 0; k < 3; ++k)
        for (int fq = 0; fq < 2; ++fq) {
          const float s = (float)(1 << fq);
          const int j = k * 2 + fq;
          bufC[p * SC + 294 + j] = (h16)__sinf(pc[k] * s);
          bufC[p * SC + 300 + j] = (h16)__cosf(pc[k] * s);
          bufC[p * SC + 306 + j] = (h16)__sinf(vc[k] * s);
          bufC[p * SC + 312 + j] = (h16)__cosf(vc[k] * s);
        }
      bufC[p * SC + 318] = (h16)0.0f;
      bufC[p * SC + 319] = (h16)0.0f;
    }
  }
  __syncthreads();

  // ---- Density MLP: 64 -> 256 (relu) -> 256 (linear) -> 272 (257 valid) ----
  wave_gemm(bufA, SA, bufB, SB, wp + OFF_DW1, db1, 64, 256, 256, true, tp, lane);
  __syncthreads();
  wave_gemm(bufB, SB, bufA, SA, wp + OFF_DW2, db2, 256, 256, 256, false, tp, lane);
  __syncthreads();
  wave_gemm(bufA, SA, bufB, SB, wp + OFF_DW3, db3, 256, 272, 257, false, tp, lane);
  __syncthreads();

  // ---- Extract sigma_mlp, copy dfeat -> cMLP input, build tiny-MLP input ----
  {
    const int p  = tp + l16;
    const int h0 = (lane >> 4) * 128;
    for (int j = 0; j < 128; ++j)
      bufC[p * SC + 3 + h0 + j] = bufB[p * SB + 1 + h0 + j];
    if (lane < 16) {
      const float sm = (float)bufB[p * SB + 0];
      sigmaL[p] = sm;
      bufA[p * SA + 0] = (h16)tsdfL[p];
      bufA[p * SA + 1] = (h16)sm;
      for (int j = 2; j < 32; ++j) bufA[p * SA + j] = (h16)0.0f;
    }
  }
  __syncthreads();

  // ---- Tiny TSDF-blend MLP: 32 -> 64 -> 128 -> 128 -> 64 -> 16 (2 valid) ----
  wave_gemm(bufA, SA, bufB, SB, wp + OFF_TW1, tb1, 32, 64, 64, true, tp, lane);
  __syncthreads();
  wave_gemm(bufB, SB, bufA, SA, wp + OFF_TW2, tb2, 64, 128, 128, true, tp, lane);
  __syncthreads();
  wave_gemm(bufA, SA, bufB, SB, wp + OFF_TW3, tb3, 128, 128, 128, true, tp, lane);
  __syncthreads();
  wave_gemm(bufB, SB, bufA, SA, wp + OFF_TW4, tb4, 128, 64, 64, true, tp, lane);
  __syncthreads();
  wave_gemm(bufA, SA, bufB, SB, wp + OFF_TW5, tb5, 64, 16, 2, false, tp, lane);
  __syncthreads();

  // ---- softmax blend -> sigma (pre-softplus) ----
  if (lane < 16) {
    const int p  = tp + l16;
    const int gp = blockIdx.x * 32 + p;
    const float t0 = (float)bufB[p * SB + 0];
    const float t1 = (float)bufB[p * SB + 1];
    const float m  = fmaxf(t0, t1);
    const float e0 = __expf(t0 - m), e1 = __expf(t1 - m);
    const float w0 = e0 / (e0 + e1), w1 = e1 / (e0 + e1);
    sigOut[gp] = w0 * sigmaL[p] + w1 * (1.0f - fabsf(tsdfL[p]));
  }
  __syncthreads();

  // ---- Color MLP: 320 -> 256 (relu) -> 256 (relu) -> 16 (3 valid) ----
  wave_gemm(bufC, SC, bufA, SA, wp + OFF_CW1, cb1, 320, 256, 256, true, tp, lane);
  __syncthreads();
  wave_gemm(bufA, SA, bufB, SB, wp + OFF_CW2, cb2, 256, 256, 256, true, tp, lane);
  __syncthreads();
  wave_gemm(bufB, SB, bufA, SA, wp + OFF_CW3, cb3, 256, 16, 3, false, tp, lane);
  __syncthreads();

  if (lane < 16) {
    const int p  = tp + l16;
    const int gp = blockIdx.x * 32 + p;
#pragma unroll
    for (int i = 0; i < 3; ++i) {
      const float x = (float)bufA[p * SA + i];
      rgbOut[gp * 3 + i] = 1.0f / (1.0f + __expf(-x));
    }
  }
}

// ---------------- per-ray compositing ----------------------------------------
__global__ void composite_kernel(const float* __restrict__ sig,
                                 const float* __restrict__ rgb,
                                 const float* __restrict__ dists,
                                 float* __restrict__ out) {
  const int r = blockIdx.x * blockDim.x + threadIdx.x;
  if (r >= 2048) return;
  float T = 1.0f, a0 = 0.0f, a1 = 0.0f, a2 = 0.0f;
  for (int s = 0; s < 128; ++s) {
    const int idx = r * 128 + s;
    const float x  = sig[idx];
    const float sp = (x > 20.0f) ? x : log1pf(__expf(x));
    const float al = 1.0f - __expf(-sp * dists[idx]);
    const float w  = al * T;
    a0 += w * rgb[idx * 3 + 0];
    a1 += w * rgb[idx * 3 + 1];
    a2 += w * rgb[idx * 3 + 2];
    T *= (1.0f - al + 1e-10f);
  }
  out[r * 3 + 0] = a0;
  out[r * 3 + 1] = a1;
  out[r * 3 + 2] = a2;
}

// ---------------- launch ------------------------------------------------------
extern "C" void kernel_launch(void* const* d_in, const int* in_sizes, int n_in,
                              void* d_out, int out_size, void* d_ws, size_t ws_size,
                              hipStream_t stream) {
  (void)in_sizes; (void)n_in; (void)out_size; (void)ws_size;
  const float* xyz       = (const float*)d_in[0];
  const float* viewdirs  = (const float*)d_in[1];
  const float* dists     = (const float*)d_in[2];
  const float* tsdf_grid = (const float*)d_in[3];
  const float* feat_grid = (const float*)d_in[4];
  const float* colr_grid = (const float*)d_in[5];
  const float* dw1 = (const float*)d_in[6];  const float* db1 = (const float*)d_in[7];
  const float* dw2 = (const float*)d_in[8];  const float* db2 = (const float*)d_in[9];
  const float* dw3 = (const float*)d_in[10]; const float* db3 = (const float*)d_in[11];
  const float* cw1 = (const float*)d_in[12]; const float* cb1 = (const float*)d_in[13];
  const float* cw2 = (const float*)d_in[14]; const float* cb2 = (const float*)d_in[15];
  const float* cw3 = (const float*)d_in[16]; const float* cb3 = (const float*)d_in[17];
  const float* tw1 = (const float*)d_in[18]; const float* tb1 = (const float*)d_in[19];
  const float* tw2 = (const float*)d_in[20]; const float* tb2 = (const float*)d_in[21];
  const float* tw3 = (const float*)d_in[22]; const float* tb3 = (const float*)d_in[23];
  const float* tw4 = (const float*)d_in[24]; const float* tb4 = (const float*)d_in[25];
  const float* tw5 = (const float*)d_in[26]; const float* tb5 = (const float*)d_in[27];

  h16*   wpack  = (h16*)d_ws;
  float* sigArr = (float*)((char*)d_ws + (1u << 20));
  float* rgbArr = (float*)((char*)d_ws + (2u << 20));

  PackArgs pa;
  pa.d[0]  = { dw1, OFF_DW1, 256,  35, 256,  64 };
  pa.d[1]  = { dw2, OFF_DW2, 256, 256, 256, 256 };
  pa.d[2]  = { dw3, OFF_DW3, 257, 256, 272, 256 };
  pa.d[3]  = { cw1, OFF_CW1, 256, 318, 256, 320 };
  pa.d[4]  = { cw2, OFF_CW2, 256, 256, 256, 256 };
  pa.d[5]  = { cw3, OFF_CW3,   3, 256,  16, 256 };
  pa.d[6]  = { tw1, OFF_TW1,  64,   2,  64,  32 };
  pa.d[7]  = { tw2, OFF_TW2, 128,  64, 128,  64 };
  pa.d[8]  = { tw3, OFF_TW3, 128, 128, 128, 128 };
  pa.d[9]  = { tw4, OFF_TW4,  64, 128,  64, 128 };
  pa.d[10] = { tw5, OFF_TW5,   2,  64,  16,  64 };

  pack_weights_kernel<<<(WPACK_TOTAL + 255) / 256, 256, 0, stream>>>(pa, wpack);

  // 262144 points, 32 points per block
  tsdfvol_fused_kernel<<<262144 / 32, 64, 0, stream>>>(
      xyz, viewdirs, tsdf_grid, feat_grid, colr_grid, wpack,
      db1, db2, db3, cb1, cb2, cb3, tb1, tb2, tb3, tb4, tb5,
      sigArr, rgbArr);

  composite_kernel<<<(2048 + 127) / 128, 128, 0, stream>>>(sigArr, rgbArr, dists,
                                                           (float*)d_out);
}
